// TurboQuantLinear_1271310320439
// MI455X (gfx1250) — matrix operand order repaired and checked by reading, not compiled, not run on hardware
//
#include <hip/hip_runtime.h>
#include <hip/hip_bf16.h>
#include <stdint.h>

typedef __attribute__((ext_vector_type(16))) _Float16 v16h;
typedef __attribute__((ext_vector_type(8)))  _Float16 v8h;
typedef __attribute__((ext_vector_type(8)))  float    v8f;

union FragU { v16h full; v8h half8[2]; };

// ---------------------------------------------------------------------------
// Tiling: 128 threads = 4 wave32 waves, arranged 2 (token) x 2 (out).
// Block tile: 64 tokens x 128 outputs x 64 K. Per-wave: 32x64 = 2x4 WMMA accs
// => 8 WMMA per 32-K chunk vs 12 ds_load_b128 (fragment reuse).
// LDS rows padded to 72 halfs (36 dwords) -> conflict-free bank phases.
// Tile fill uses CDNA5 async global->LDS copies (ASYNCcnt), double-buffered.
// ---------------------------------------------------------------------------
constexpr int BT  = 64;
constexpr int BO  = 128;
constexpr int BK  = 64;
constexpr int LDA = BK + 8;   // padded LDS row stride in halfs
constexpr int NTH = 128;

__device__ __forceinline__ void async_copy_b128(const _Float16* gptr, _Float16* lptr)
{
    // CDNA5 GLOBAL_LOAD_ASYNC_TO_LDS_B128 (GV mode): per-lane 16B global -> LDS.
    // Low 32 bits of a generic shared pointer are the LDS byte address.
    unsigned lds = (unsigned)(uintptr_t)lptr;
    asm volatile("global_load_async_to_lds_b128 %0, %1, off"
                 :: "v"(lds), "v"(gptr) : "memory");
}

__device__ __forceinline__ void wait_async0()
{
#if __has_builtin(__builtin_amdgcn_s_wait_asynccnt)
    __builtin_amdgcn_s_wait_asynccnt(0);
#else
    asm volatile("s_wait_asynccnt 0x0" ::: "memory");
#endif
}

template <bool HALF_OUT, bool ADD>
__global__ __launch_bounds__(NTH)
void tql_wmma_gemm(const _Float16* __restrict__ A,   // Tdim x K row-major (f16)
                   const _Float16* __restrict__ B,   // Mout x K row-major (f16), used as B^T
                   void* __restrict__ Cout,          // Tdim x Mout (f16 if HALF_OUT else f32)
                   int K, int Mout)
{
    __shared__ _Float16 sA[2][BT * LDA];
    __shared__ _Float16 sB[2][BO * LDA];

    const int tid   = threadIdx.x;
    const int lane  = tid & 31;
    const int wave  = tid >> 5;
    const int waveT = wave & 1;   // token direction (2 waves x 32 tokens)
    const int waveO = wave >> 1;  // output direction (2 waves x 64 outputs)

    const int t0 = blockIdx.y * BT;
    const int m0 = blockIdx.x * BO;

    // Async tile fill: A tile 64x64 halfs (4 sweeps), B tile 128x64 (8 sweeps).
    auto fill_tile = [&](int kt, int buf) {
        const int k0 = kt * BK;
#pragma unroll
        for (int s = 0; s < 4; ++s) {
            int h = (s * NTH + tid) * 8;
            int r = h / BK, c = h % BK;
            async_copy_b128(A + (size_t)(t0 + r) * K + k0 + c, &sA[buf][r * LDA + c]);
        }
#pragma unroll
        for (int s = 0; s < 8; ++s) {
            int h = (s * NTH + tid) * 8;
            int r = h / BK, c = h % BK;
            async_copy_b128(B + (size_t)(m0 + r) * K + k0 + c, &sB[buf][r * LDA + c]);
        }
    };

    v8f acc[2][4];
#pragma unroll
    for (int i = 0; i < 2; ++i)
#pragma unroll
        for (int j = 0; j < 4; ++j)
            acc[i][j] = v8f{};

    const int NK = K / BK;
    fill_tile(0, 0);
    wait_async0();
    __syncthreads();

    for (int kt = 0; kt < NK; ++kt) {
        const int cur = kt & 1;
        if (kt + 1 < NK) fill_tile(kt + 1, cur ^ 1);

#pragma unroll
        for (int kc = 0; kc < BK / 32; ++kc) {
            // A fragment (16x32 f16): lane holds row (lane&15); K in 8-wide
            // chunks split across lane halves per ISA 7.12.2.
            FragU a[2], b[4];
            const int arow = lane & 15;
            const int akb  = (lane >> 4) * 8;
#pragma unroll
            for (int i = 0; i < 2; ++i) {
                const int row = waveT * 32 + i * 16 + arow;
                const _Float16* p = &sA[cur][row * LDA + kc * 32];
                a[i].half8[0] = *(const v8h*)(p + akb);
                a[i].half8[1] = *(const v8h*)(p + akb + 16);
            }
            // B fragment (32x16 f16): lane n holds output row (m-tile + (lane&15));
            // K in 8-wide chunks split across lane halves, same order as A.
            const int boff = (lane >> 4) * 8;
#pragma unroll
            for (int j = 0; j < 4; ++j) {
                const int row = waveO * 64 + j * 16 + (lane & 15);
                const _Float16* p = &sB[cur][row * LDA + kc * 32 + boff];
                b[j].half8[0] = *(const v8h*)(p);
                b[j].half8[1] = *(const v8h*)(p + 16);
            }
#pragma unroll
            for (int i = 0; i < 2; ++i)
#pragma unroll
                for (int j = 0; j < 4; ++j)
                    acc[i][j] = __builtin_amdgcn_wmma_f32_16x16x32_f16(
                        false, a[i].full, false, b[j].full,
                        (short)0, acc[i][j], false, false);
        }
        if (kt + 1 < NK) wait_async0();   // own async fills landed in LDS
        __syncthreads();                  // everyone's fills visible
    }

    // Epilogue: C/D layout — VGPR v, lanes 0-15: M=v, lanes 16-31: M=v+8.
    const int col   = lane & 15;
    const int rbase = (lane >> 4) * 8;
#pragma unroll
    for (int i = 0; i < 2; ++i) {
#pragma unroll
        for (int j = 0; j < 4; ++j) {
            const int trow0 = t0 + waveT * 32 + i * 16;
            const int mcol  = m0 + waveO * 64 + j * 16 + col;
#pragma unroll
            for (int v = 0; v < 8; ++v) {
                const int t = trow0 + rbase + v;
                const size_t off = (size_t)t * Mout + mcol;
                const float val = acc[i][j][v];
                if (HALF_OUT) {
                    ((_Float16*)Cout)[off] = (_Float16)val;
                } else {
                    float* o = (float*)Cout;
                    if (ADD) o[off] += val; else o[off] = val;
                }
            }
        }
    }
}

// ---------------------------------------------------------------------------
// Dequant: unpack 4-bit indices, codebook gather (LDS-resident), per-group
// norm scale (GROUP=128, scale = norm / sqrt(128)), emit f16 pairs.
// ---------------------------------------------------------------------------
__global__ __launch_bounds__(256)
void tql_dequant(const int* __restrict__ packed,     // Mrows x halfN (byte values)
                 const float* __restrict__ norms,    // Mrows x groupsPerRow
                 const float* __restrict__ codebook, // 16
                 _Float16* __restrict__ Wh,          // Mrows x (2*halfN)
                 int halfN, int groupsPerRow, size_t total)
{
    __shared__ float cb[16];
    if (threadIdx.x < 16) cb[threadIdx.x] = codebook[threadIdx.x];
    __syncthreads();

    const float inv_scale = 0.08838834764831845f;  // 1/sqrt(128)
    size_t gid = (size_t)blockIdx.x * blockDim.x + threadIdx.x;
    if (gid >= total) return;

    const int m = (int)(gid / halfN);
    const int j = (int)(gid % halfN);
    const int v = packed[gid];
    const int n0 = 2 * j;
    const int grp = n0 >> 7;  // GROUP = 128
    const float s = norms[(size_t)m * groupsPerRow + grp] * inv_scale;

    union { _Float16 h[2]; unsigned u; } pk;
    pk.h[0] = (_Float16)(cb[v & 15] * s);          // low nibble -> even column
    pk.h[1] = (_Float16)(cb[(v >> 4) & 15] * s);   // high nibble -> odd column
    *(unsigned*)(&Wh[(size_t)m * (2 * (size_t)halfN) + n0]) = pk.u;
}

// f32 -> f16 elementwise, 4-wide (all sizes here are multiples of 4)
__global__ __launch_bounds__(256)
void tql_cvt_f32_f16(const float* __restrict__ in, _Float16* __restrict__ out, size_t n)
{
    size_t i = ((size_t)blockIdx.x * blockDim.x + threadIdx.x) * 4;
    if (i + 3 < n) {
        float4 f = *(const float4*)(in + i);
        union { _Float16 h[4]; unsigned long long u; } pk;
        pk.h[0] = (_Float16)f.x; pk.h[1] = (_Float16)f.y;
        pk.h[2] = (_Float16)f.z; pk.h[3] = (_Float16)f.w;
        *(unsigned long long*)(out + i) = pk.u;
    }
}

extern "C" void kernel_launch(void* const* d_in, const int* in_sizes, int n_in,
                              void* d_out, int out_size, void* d_ws, size_t ws_size,
                              hipStream_t stream)
{
    (void)in_sizes; (void)n_in; (void)out_size; (void)ws_size;

    constexpr int T = 512, N = 4096, M = 4096;

    const float* x    = (const float*)d_in[0];
    const float* Pi   = (const float*)d_in[1];
    const int*   idx1 = (const int*)d_in[2];
    const float* nrm1 = (const float*)d_in[3];
    const float* cb1  = (const float*)d_in[4];
    const int*   idx2 = (const int*)d_in[5];
    const float* nrm2 = (const float*)d_in[6];
    const float* cb2  = (const float*)d_in[7];
    const float* Pi2  = (const float*)d_in[8];
    float* out = (float*)d_out;

    // Workspace layout: xh(4MB) | Bh(32MB, reused for Pi/Pi2/W1/W2) | Y1h | Y2h
    char* ws = (char*)d_ws;
    _Float16* xh  = (_Float16*)ws;  ws += (size_t)T * N * sizeof(_Float16);
    _Float16* Bh  = (_Float16*)ws;  ws += (size_t)M * N * sizeof(_Float16);
    _Float16* Y1h = (_Float16*)ws;  ws += (size_t)T * M * sizeof(_Float16);
    _Float16* Y2h = (_Float16*)ws;

    const dim3 gemm_grid(M / BO, T / BT);   // 32 x 8 = 256 workgroups
    const int cvt_big   = (M * N) / 4 / 256;
    const int cvt_small = (T * N) / 4 / 256;
    const size_t dq_total = (size_t)M * (N / 2);
    const int dq_blocks = (int)(dq_total / 256);

    // Stage 0: x -> f16 (kept resident for both passes)
    tql_cvt_f32_f16<<<cvt_small, 256, 0, stream>>>(x, xh, (size_t)T * N);

    // Pass 1, rotation GEMM: Y1 = x @ Pi^T  (f16 out)
    tql_cvt_f32_f16<<<cvt_big, 256, 0, stream>>>(Pi, Bh, (size_t)M * N);
    tql_wmma_gemm<true, false><<<gemm_grid, NTH, 0, stream>>>(xh, Bh, Y1h, N, M);

    // Pass 2, rotation GEMM: Y2 = x @ Pi2^T  (f16 out, reuse Bh)
    tql_cvt_f32_f16<<<cvt_big, 256, 0, stream>>>(Pi2, Bh, (size_t)M * N);
    tql_wmma_gemm<true, false><<<gemm_grid, NTH, 0, stream>>>(xh, Bh, Y2h, N, M);

    // Pass 1, weight GEMM: out = Y1 @ W1^T  (f32 overwrite)
    tql_dequant<<<dq_blocks, 256, 0, stream>>>(idx1, nrm1, cb1, Bh, N / 2, N / 128, dq_total);
    tql_wmma_gemm<false, false><<<gemm_grid, NTH, 0, stream>>>(Y1h, Bh, (void*)out, N, M);

    // Pass 2, weight GEMM: out += Y2 @ W2^T
    tql_dequant<<<dq_blocks, 256, 0, stream>>>(idx2, nrm2, cb2, Bh, N / 2, N / 128, dq_total);
    tql_wmma_gemm<false, true><<<gemm_grid, NTH, 0, stream>>>(Y2h, Bh, (void*)out, N, M);
}
